// AttentionBlock_67662914781911
// MI455X (gfx1250) — matrix operand order repaired and checked by reading, not compiled, not run on hardware
//
#include <hip/hip_runtime.h>

typedef _Float16 v16h __attribute__((ext_vector_type(16)));
typedef _Float16 v8h  __attribute__((ext_vector_type(8)));
typedef _Float16 v4h  __attribute__((ext_vector_type(4)));
typedef float    v8f  __attribute__((ext_vector_type(8)));
typedef float    v4f  __attribute__((ext_vector_type(4)));

#define N_TOK  16384      // B*H*W
#define CDIM   512
#define NSEQ   4096       // H*W
#define QBLK   1024       // N_TOK / 16
#define CHUNK  1024       // keys per flash chunk
#define NCHUNK 4

__device__ __forceinline__ v8f wmma_f16(v16h a, v16h b, v8f c) {
  // D = A(16x32 f16) x B(32x16 f16) + C(16x16 f32)
  return __builtin_amdgcn_wmma_f32_16x16x32_f16(false, a, false, b, (short)0, c,
                                                false, false);
}

__device__ __forceinline__ v8f zero8() {
  v8f z;
#pragma unroll
  for (int i = 0; i < 8; ++i) z[i] = 0.f;
  return z;
}

// A-matrix fragment (16x32 f16, row = lane&15):
// lanes 0-15 hold K {0..15}, lanes 16-31 hold K {16..31}
__device__ __forceinline__ v16h load_afrag(const _Float16* base, int hi) {
  v8h lo = *(const v8h*)(base + hi * 16);
  v8h hh = *(const v8h*)(base + hi * 16 + 8);
  v16h r;
#pragma unroll
  for (int i = 0; i < 8; ++i) { r[i] = lo[i]; r[i + 8] = hh[i]; }
  return r;
}

// B-matrix fragment (32x16 f16, col = lane&15): 16 contiguous halves along K,
// caller passes base already offset by (hi*16) K-elements.
__device__ __forceinline__ v16h load_bfrag(const _Float16* base) {
  v8h lo = *(const v8h*)(base);
  v8h hh = *(const v8h*)(base + 8);
  v16h r;
#pragma unroll
  for (int i = 0; i < 8; ++i) { r[i] = lo[i]; r[i + 8] = hh[i]; }
  return r;
}

// ---------------- weight transpose + f16 convert: Wt[d][c] = W[c][d] ----------
__global__ __launch_bounds__(256) void wt_kernel(const float* __restrict__ w,
                                                 _Float16* __restrict__ wt) {
  int gid = blockIdx.x * 256 + threadIdx.x;   // 512*512 threads
  int c = gid & (CDIM - 1);
  int d = gid >> 9;
  wt[(size_t)d * CDIM + c] = (_Float16)w[(size_t)c * CDIM + d];
}

// ---------------- GroupNorm stats: one block per (batch, group) --------------
__global__ __launch_bounds__(256) void gn_stats_kernel(const float* __restrict__ x,
                                                       float* __restrict__ stats) {
  int bg = blockIdx.x;                 // 0..127
  int b = bg >> 5, g = bg & 31;
  const float* base = x + (size_t)b * NSEQ * CDIM + g * 16;
  float s = 0.f, s2 = 0.f;
  for (int p = threadIdx.x; p < NSEQ; p += 256) {
    const float* px = base + (size_t)p * CDIM;
#pragma unroll
    for (int c = 0; c < 16; c += 4) {
      v4f v = *(const v4f*)(px + c);
#pragma unroll
      for (int k = 0; k < 4; ++k) { s += v[k]; s2 += v[k] * v[k]; }
    }
  }
#pragma unroll
  for (int m = 16; m >= 1; m >>= 1) {
    s  += __shfl_xor(s, m, 32);
    s2 += __shfl_xor(s2, m, 32);
  }
  __shared__ float red[16];
  int wv = threadIdx.x >> 5;
  if ((threadIdx.x & 31) == 0) { red[wv * 2] = s; red[wv * 2 + 1] = s2; }
  __syncthreads();
  if (threadIdx.x == 0) {
    float S = 0.f, S2 = 0.f;
#pragma unroll
    for (int i = 0; i < 8; ++i) { S += red[2 * i]; S2 += red[2 * i + 1]; }
    const float inv = 1.0f / 65536.0f;
    float mean = S * inv;
    float var  = S2 * inv - mean * mean;
    stats[2 * bg]     = mean;
    stats[2 * bg + 1] = rsqrtf(var + 1e-5f);
  }
}

// ---------------- GroupNorm apply + affine -> f16 ----------------------------
__global__ __launch_bounds__(256) void gn_norm_kernel(
    const float* __restrict__ x, const float* __restrict__ gamma,
    const float* __restrict__ beta, const float* __restrict__ stats,
    _Float16* __restrict__ xh) {
  int gid = blockIdx.x * 256 + threadIdx.x;  // N_TOK*CDIM/4 threads
  int row = gid >> 7;
  int c = (gid & 127) << 2;
  int b = row >> 12, g = c >> 4;
  float mean = stats[2 * ((b << 5) + g)];
  float rstd = stats[2 * ((b << 5) + g) + 1];
  v4f v = *(const v4f*)(x + (size_t)row * CDIM + c);
  v4h pk;
#pragma unroll
  for (int k = 0; k < 4; ++k)
    pk[k] = (_Float16)(((v[k] - mean) * rstd) * gamma[c + k] + beta[c + k]);
  *(v4h*)(xh + (size_t)row * CDIM + c) = pk;
}

// ---------------- WMMA GEMM: out[n,d] = A[n,:] . Wt[d,:] + bias --------------
// MODE 0: f16 row-major out (scaled)     -> Q / K
// MODE 1: f16 transposed out [b][d][n]   -> V
// MODE 2: f32 out + bias + residual      -> final projection
template <int MODE>
__global__ __launch_bounds__(128) void gemm_kernel(
    const _Float16* __restrict__ A, const _Float16* __restrict__ Bt,
    const float* __restrict__ bias, float scale, _Float16* __restrict__ outh,
    float* __restrict__ outf, const float* __restrict__ residual) {
  int lane = threadIdx.x & 31;
  int wave = threadIdx.x >> 5;
  int colL = lane & 15;
  int hi = lane >> 4;
  int rt = blockIdx.x * 4 + wave;          // 16-row tile
  int cbase = blockIdx.y * 64;             // 64-col slab
  v8f acc[4];
#pragma unroll
  for (int j = 0; j < 4; ++j) acc[j] = zero8();
  const _Float16* ap = A + (size_t)(rt * 16 + colL) * CDIM;
#pragma unroll 4
  for (int kk = 0; kk < 16; ++kk) {
    v16h a = load_afrag(ap + kk * 32, hi);
#pragma unroll
    for (int j = 0; j < 4; ++j) {
      const _Float16* bp =
          Bt + (size_t)(cbase + j * 16 + colL) * CDIM + kk * 32 + hi * 16;
      acc[j] = wmma_f16(a, load_bfrag(bp), acc[j]);
    }
  }
#pragma unroll
  for (int j = 0; j < 4; ++j) {
    int d = cbase + j * 16 + colL;
    float bv = bias[d];
    if (MODE == 1) {
      int rowbase = rt * 16;
      int b = rowbase >> 12;
      int nloc = (rowbase & (NSEQ - 1)) + 8 * hi;
      v8h pk;
#pragma unroll
      for (int r = 0; r < 8; ++r) pk[r] = (_Float16)((acc[j][r] + bv) * scale);
      *(v8h*)(outh + ((size_t)(b * CDIM + d)) * NSEQ + nloc) = pk;
    } else {
#pragma unroll
      for (int r = 0; r < 8; ++r) {
        size_t row = (size_t)(rt * 16 + r + 8 * hi);
        float v = (acc[j][r] + bv) * scale;
        if (MODE == 0) outh[row * CDIM + d] = (_Float16)v;
        else           outf[row * CDIM + d] = v + residual[row * CDIM + d];
      }
    }
  }
}

// ---------------- Flash attention: 1 wave = 16 query rows --------------------
// LDS: sQ 16KB (Q tile, A-fragments) + sP 32KB (scores/P) + s_m = 48KB
__global__ __launch_bounds__(32) void flash_kernel(
    const _Float16* __restrict__ qh, const _Float16* __restrict__ kh,
    const _Float16* __restrict__ vt, float* __restrict__ osc,
    _Float16* __restrict__ attn) {
  __shared__ _Float16 sQ[16 * CDIM];       // 16 KB
  __shared__ _Float16 sP[16 * CHUNK];      // 32 KB raw scores -> P
  __shared__ float    s_m[16];
  int lane = threadIdx.x;
  int colL = lane & 15;
  int hi = lane >> 4;
  int blk = blockIdx.x;                    // 0..1023
  int b = blk >> 8;
  int qrow = blk * 16;
  int kRow0 = b * NSEQ;
  size_t vtBase = (size_t)b * CDIM * NSEQ;
  size_t oBase = (size_t)blk * 16 * CDIM;

  // stage Q tile (rows qrow..qrow+15) into LDS, coalesced b128 copies
  {
    const _Float16* qsrc = qh + (size_t)qrow * CDIM;
#pragma unroll
    for (int i = 0; i < (16 * CDIM) / (32 * 8); ++i) {
      int off = (lane + i * 32) * 8;
      *(v8h*)(sQ + off) = *(const v8h*)(qsrc + off);
    }
  }
  __syncthreads();

  const _Float16* aq = sQ + colL * CDIM;   // A-fragment base (row = lane&15)
  float m_run[8], l_run[8];
#pragma unroll
  for (int r = 0; r < 8; ++r) { m_run[r] = -1e30f; l_run[r] = 0.f; }

  v16h ones;
#pragma unroll
  for (int i = 0; i < 16; ++i) ones[i] = (_Float16)1.0f;

  for (int ch = 0; ch < NCHUNK; ++ch) {
    int kbase = ch * CHUNK;
    // ---- phase 1: S = Q.K^T (scale pre-folded into Q) -> LDS raw, lane max
    float mloc[8];
#pragma unroll
    for (int r = 0; r < 8; ++r) mloc[r] = -1e30f;
    for (int kt = 0; kt < CHUNK / 32; ++kt) {
      v8f s0 = zero8(), s1 = zero8();
      const _Float16* kp0 =
          kh + (size_t)(kRow0 + kbase + kt * 32 + colL) * CDIM;
      const _Float16* kp1 = kp0 + (size_t)16 * CDIM;
#pragma unroll 4
      for (int kk = 0; kk < 16; ++kk) {
        v16h a = load_afrag(aq + kk * 32, hi);
        s0 = wmma_f16(a, load_bfrag(kp0 + kk * 32 + hi * 16), s0);
        s1 = wmma_f16(a, load_bfrag(kp1 + kk * 32 + hi * 16), s1);
      }
#pragma unroll
      for (int r = 0; r < 8; ++r) {
        mloc[r] = fmaxf(mloc[r], fmaxf(s0[r], s1[r]));
        sP[(r + 8 * hi) * CHUNK + kt * 32 + colL]      = (_Float16)s0[r];
        sP[(r + 8 * hi) * CHUNK + kt * 32 + 16 + colL] = (_Float16)s1[r];
      }
    }
    // row max across the 16-lane column group
#pragma unroll
    for (int m = 1; m <= 8; m <<= 1)
#pragma unroll
      for (int r = 0; r < 8; ++r)
        mloc[r] = fmaxf(mloc[r], __shfl_xor(mloc[r], m, 32));
    float alpha[8];
#pragma unroll
    for (int r = 0; r < 8; ++r) {
      float mn = fmaxf(m_run[r], mloc[r]);
      alpha[r] = __expf(m_run[r] - mn);
      m_run[r] = mn;
    }
    if (lane == 0)
#pragma unroll
      for (int r = 0; r < 8; ++r) s_m[r] = m_run[r];
    if (lane == 16)
#pragma unroll
      for (int r = 0; r < 8; ++r) s_m[8 + r] = m_run[r];
    __syncthreads();
    // ---- vectorized exp in place: 8 halves per DS op
    for (int i = lane; i < (16 * CHUNK) / 8; i += 32) {
      int off = i * 8;
      float mr = s_m[off >> 10];           // 1024 halves per row
      v8h pv = *(v8h*)(sP + off);
      v8h ev;
#pragma unroll
      for (int k = 0; k < 8; ++k) ev[k] = (_Float16)__expf((float)pv[k] - mr);
      *(v8h*)(sP + off) = ev;
    }
    __syncthreads();
    // ---- P.V  (l computed via WMMA against a ones B-matrix: C/D layout)
    v8f accl = zero8();
    for (int cc = 0; cc < 4; ++cc) {
      int cb = cc * 128;
      v8f acc[8];
#pragma unroll
      for (int ct = 0; ct < 8; ++ct) {
        if (ch == 0) acc[ct] = zero8();
        else {
          v8f t;
#pragma unroll
          for (int r = 0; r < 8; ++r)
            t[r] = osc[oBase + (size_t)(r + 8 * hi) * CDIM + cb + ct * 16 + colL] *
                   alpha[r];
          acc[ct] = t;
        }
      }
      for (int ks = 0; ks < CHUNK / 32; ++ks) {
        v16h pa = load_afrag(&sP[colL * CHUNK + ks * 32], hi);
#pragma unroll
        for (int ct = 0; ct < 8; ++ct) {
          const _Float16* vp = vt + vtBase +
                               (size_t)(cb + ct * 16 + colL) * NSEQ + kbase +
                               ks * 32 + hi * 16;
          acc[ct] = wmma_f16(pa, load_bfrag(vp), acc[ct]);
        }
        if (cc == 0) accl = wmma_f16(pa, ones, accl);
      }
#pragma unroll
      for (int ct = 0; ct < 8; ++ct)
#pragma unroll
        for (int r = 0; r < 8; ++r)
          osc[oBase + (size_t)(r + 8 * hi) * CDIM + cb + ct * 16 + colL] =
              acc[ct][r];
    }
#pragma unroll
    for (int r = 0; r < 8; ++r) l_run[r] = l_run[r] * alpha[r] + accl[r];
    __syncthreads();
  }
  // ---- normalize and emit f16 attention output
  float invl[8];
#pragma unroll
  for (int r = 0; r < 8; ++r) invl[r] = 1.0f / l_run[r];
  for (int ct = 0; ct < 32; ++ct)
#pragma unroll
    for (int r = 0; r < 8; ++r) {
      float v =
          osc[oBase + (size_t)(r + 8 * hi) * CDIM + ct * 16 + colL] * invl[r];
      attn[(size_t)(qrow + r + 8 * hi) * CDIM + ct * 16 + colL] = (_Float16)v;
    }
}

// ---------------------------------------------------------------------------
extern "C" void kernel_launch(void* const* d_in, const int* in_sizes, int n_in,
                              void* d_out, int out_size, void* d_ws,
                              size_t ws_size, hipStream_t stream) {
  (void)in_sizes; (void)n_in; (void)out_size; (void)ws_size;
  const float* x     = (const float*)d_in[0];
  const float* gamma = (const float*)d_in[1];
  const float* beta  = (const float*)d_in[2];
  const float* wq    = (const float*)d_in[3];
  const float* bq    = (const float*)d_in[4];
  const float* wk    = (const float*)d_in[5];
  const float* bk    = (const float*)d_in[6];
  const float* wv    = (const float*)d_in[7];
  const float* bv    = (const float*)d_in[8];
  const float* wo    = (const float*)d_in[9];
  const float* bo    = (const float*)d_in[10];
  float* out = (float*)d_out;

  char* w = (char*)d_ws;
  float* stats   = (float*)w;   w += 4096;
  _Float16* wqt  = (_Float16*)w; w += (size_t)CDIM * CDIM * 2;
  _Float16* wkt  = (_Float16*)w; w += (size_t)CDIM * CDIM * 2;
  _Float16* wvt  = (_Float16*)w; w += (size_t)CDIM * CDIM * 2;
  _Float16* wot  = (_Float16*)w; w += (size_t)CDIM * CDIM * 2;
  _Float16* xh   = (_Float16*)w; w += (size_t)N_TOK * CDIM * 2;
  _Float16* qhp  = (_Float16*)w; w += (size_t)N_TOK * CDIM * 2;
  _Float16* khp  = (_Float16*)w; w += (size_t)N_TOK * CDIM * 2;
  _Float16* vtp  = (_Float16*)w; w += (size_t)N_TOK * CDIM * 2;
  float* osc     = (float*)w;    w += (size_t)N_TOK * CDIM * 4;
  _Float16* attn = xh;  // xh is dead after the QKV GEMMs -> reuse

  wt_kernel<<<1024, 256, 0, stream>>>(wq, wqt);
  wt_kernel<<<1024, 256, 0, stream>>>(wk, wkt);
  wt_kernel<<<1024, 256, 0, stream>>>(wv, wvt);
  wt_kernel<<<1024, 256, 0, stream>>>(wo, wot);

  gn_stats_kernel<<<128, 256, 0, stream>>>(x, stats);
  gn_norm_kernel<<<(N_TOK * CDIM / 4) / 256, 256, 0, stream>>>(x, gamma, beta,
                                                               stats, xh);
  dim3 ggrid(N_TOK / 64, CDIM / 64);
  const float qscale = 0.044194173824159216f;  // 1/sqrt(512), folded into Q
  gemm_kernel<0><<<ggrid, 128, 0, stream>>>(xh, wqt, bq, qscale, qhp, nullptr, nullptr);
  gemm_kernel<0><<<ggrid, 128, 0, stream>>>(xh, wkt, bk, 1.0f,  khp, nullptr, nullptr);
  gemm_kernel<1><<<ggrid, 128, 0, stream>>>(xh, wvt, bv, 1.0f,  vtp, nullptr, nullptr);

  flash_kernel<<<QBLK, 32, 0, stream>>>(qhp, khp, vtp, osc, attn);

  gemm_kernel<2><<<ggrid, 128, 0, stream>>>(attn, wot, bo, 1.0f, nullptr, out, x);
}
